// EncoderBlock_11965778887066
// MI455X (gfx1250) — matrix operand order repaired and checked by reading, not compiled, not run on hardware
//
#include <hip/hip_runtime.h>
#include <math.h>

typedef __bf16 bf16;
typedef __attribute__((ext_vector_type(16))) __bf16 v16bf;
typedef __attribute__((ext_vector_type(8)))  float   v8f;

#define NB 4
#define NT 2048
#define ND 1024
#define NH 4
#define NKD 256
#define NVD 512
#define KEYD 1024
#define VALD 2048
#define MROWS (NB*NT)
#define CHUNK 64
#define VBLK 64
#define NCHUNK (NT/CHUNK)
#define NVB (NVD/VBLK)

// ---------------------------------------------------------------- WMMA helpers
__device__ inline v8f wmma_bf(v16bf a, v16bf b, v8f c) {
  return __builtin_amdgcn_wmma_f32_16x16x32_bf16(false, a, false, b, (short)0, c,
                                                 false, false);
}

// A-operand fragment (16x32, bf16): lane<16 holds K {0..7,16..23}, lane>=16 {8..15,24..31}
__device__ inline v16bf fragA_bf(const bf16* base, int stride, int row0, int kb0) {
  int lane = threadIdx.x & 31;
  const bf16* p = base + (size_t)(row0 + (lane & 15)) * stride + kb0 + ((lane >> 4) << 3);
  v16bf f;
#pragma unroll
  for (int i = 0; i < 8; ++i) { f[i] = p[i]; f[i + 8] = p[i + 16]; }
  return f;
}
__device__ inline v16bf fragA_f32(const float* base, int stride, int row0, int kb0) {
  int lane = threadIdx.x & 31;
  const float* p = base + (size_t)(row0 + (lane & 15)) * stride + kb0 + ((lane >> 4) << 3);
  v16bf f;
#pragma unroll
  for (int i = 0; i < 8; ++i) { f[i] = (bf16)p[i]; f[i + 8] = (bf16)p[i + 16]; }
  return f;
}
__device__ inline v16bf fragA_bf_scaled(const bf16* base, int stride, int row0, int kb0,
                                        const float* sc) {
  int lane = threadIdx.x & 31;
  int koff = kb0 + ((lane >> 4) << 3);
  const bf16* p = base + (size_t)(row0 + (lane & 15)) * stride + koff;
  v16bf f;
#pragma unroll
  for (int i = 0; i < 8; ++i) {
    f[i]     = (bf16)((float)p[i]      * sc[koff + i]);
    f[i + 8] = (bf16)((float)p[i + 16] * sc[koff + 16 + i]);
  }
  return f;
}
// B-operand fragment (32x16, bf16): lane&15 = N column; lane<16 K {0..7,16..23}, lane>=16 {8..15,24..31}.
// Source arrays store the B matrix transposed (row = N, contiguous K).
__device__ inline v16bf fragB_bf(const bf16* base, int stride, int col0, int kb0) {
  int lane = threadIdx.x & 31;
  const bf16* p = base + (size_t)(col0 + (lane & 15)) * stride + kb0 + ((lane >> 4) << 3);
  v16bf f;
#pragma unroll
  for (int i = 0; i < 8; ++i) { f[i] = p[i]; f[i + 8] = p[i + 16]; }
  return f;
}
__device__ inline v16bf fragB_f32(const float* base, int stride, int col0, int kb0) {
  int lane = threadIdx.x & 31;
  const float* p = base + (size_t)(col0 + (lane & 15)) * stride + kb0 + ((lane >> 4) << 3);
  v16bf f;
#pragma unroll
  for (int i = 0; i < 8; ++i) { f[i] = (bf16)p[i]; f[i + 8] = (bf16)p[i + 16]; }
  return f;
}

__device__ inline float block_sum256(float v, float* red) {
  int tid = threadIdx.x;
  red[tid] = v; __syncthreads();
  for (int s = 128; s > 0; s >>= 1) {
    if (tid < s) red[tid] += red[tid + s];
    __syncthreads();
  }
  float r = red[0]; __syncthreads();
  return r;
}

__device__ inline float sigmoidf_(float x) { return 1.f / (1.f + __expf(-x)); }
__device__ inline float siluf_(float x)    { return x * sigmoidf_(x); }

// ---------------------------------------------------- weight transpose fp32 -> bf16
__global__ void transpose_bf16_kernel(const float* __restrict__ W, bf16* __restrict__ Wt,
                                      int K, int N) {
  int total = K * N;
  for (int idx = blockIdx.x * blockDim.x + threadIdx.x; idx < total;
       idx += gridDim.x * blockDim.x) {
    int k = idx / N, n = idx - k * N;
    Wt[(size_t)n * K + k] = (bf16)W[idx];
  }
}

// ------------------------------------------------ LN1 + RMSNorm -> h (f32 + bf16)
__global__ __launch_bounds__(256) void norm1_kernel(
    const float* __restrict__ x, const float* __restrict__ g1,
    const float* __restrict__ b1, const float* __restrict__ rmsw,
    float* __restrict__ h32, bf16* __restrict__ hbf) {
  __shared__ float red[256];
  int row = blockIdx.x, tid = threadIdx.x;
  float4 xv = *(const float4*)&x[(size_t)row * ND + tid * 4];
  float xs[4] = {xv.x, xv.y, xv.z, xv.w};
  float s = xs[0] + xs[1] + xs[2] + xs[3];
  float mean = block_sum256(s, red) * (1.f / ND);
  float s2 = 0.f;
#pragma unroll
  for (int i = 0; i < 4; ++i) { float d = xs[i] - mean; s2 += d * d; }
  float var = block_sum256(s2, red) * (1.f / ND);
  float inv = rsqrtf(var + 1e-5f);
  float hs[4], s3 = 0.f;
#pragma unroll
  for (int i = 0; i < 4; ++i) {
    int c = tid * 4 + i;
    hs[i] = (xs[i] - mean) * inv * g1[c] + b1[c];
    s3 += hs[i] * hs[i];
  }
  float ms = block_sum256(s3, red) * (1.f / ND);
  float rinv = rsqrtf(ms + 1e-6f);
#pragma unroll
  for (int i = 0; i < 4; ++i) {
    int c = tid * 4 + i;
    float v = hs[i] * rinv * rmsw[c];
    h32[(size_t)row * ND + c] = v;
    hbf[(size_t)row * ND + c] = (bf16)v;
  }
}

// --------------------------------------------------------- 128x128 bf16 WMMA GEMM
// C[M,N] = A[M,K] * Bt[N,K]^T   (Bt stores B transposed, rows contiguous in K)
__global__ __launch_bounds__(256) void gemm_bf16_kernel(
    const bf16* __restrict__ A, const bf16* __restrict__ Bt,
    float* __restrict__ C, int M, int N, int K) {
  __shared__ alignas(16) bf16 As[128][32];
  __shared__ alignas(16) bf16 Bs[128][32];
  const int tid = threadIdx.x;
  const int lane = tid & 31;
  const int wid = tid >> 5;
  const int bm = blockIdx.x * 128;
  const int bn = blockIdx.y * 128;
  const int wm = (wid & 1) * 64;
  const int wn = (wid >> 1) * 32;
  v8f acc[4][2];
#pragma unroll
  for (int mt = 0; mt < 4; ++mt)
#pragma unroll
    for (int nt = 0; nt < 2; ++nt)
#pragma unroll
      for (int i = 0; i < 8; ++i) acc[mt][nt][i] = 0.f;

  for (int k0 = 0; k0 < K; k0 += 32) {
#pragma unroll
    for (int li = tid; li < 512; li += 256) {
      int row = li >> 2, seg = (li & 3) * 8;
      *(int4*)&As[row][seg] = *(const int4*)&A[(size_t)(bm + row) * K + k0 + seg];
      *(int4*)&Bs[row][seg] = *(const int4*)&Bt[(size_t)(bn + row) * K + k0 + seg];
    }
    __syncthreads();
    v16bf af[4], bfr[2];
#pragma unroll
    for (int mt = 0; mt < 4; ++mt) af[mt] = fragA_bf(&As[0][0], 32, wm + mt * 16, 0);
#pragma unroll
    for (int nt = 0; nt < 2; ++nt) bfr[nt] = fragB_bf(&Bs[0][0], 32, wn + nt * 16, 0);
#pragma unroll
    for (int mt = 0; mt < 4; ++mt)
#pragma unroll
      for (int nt = 0; nt < 2; ++nt)
        acc[mt][nt] = wmma_bf(af[mt], bfr[nt], acc[mt][nt]);
    __syncthreads();
  }
  const int hi = (lane >> 4) << 3;
  const int col = lane & 15;
#pragma unroll
  for (int mt = 0; mt < 4; ++mt)
#pragma unroll
    for (int nt = 0; nt < 2; ++nt)
#pragma unroll
      for (int v = 0; v < 8; ++v) {
        int r = bm + wm + mt * 16 + v + hi;
        int cc = bn + wn + nt * 16 + col;
        C[(size_t)r * N + cc] = acc[mt][nt][v];
      }
}

// ------------------------------------------- depthwise causal conv (CONV=4) + SiLU
__global__ void conv_silu_kernel(float* __restrict__ buf, const float* __restrict__ w,
                                 int C) {
  int gid = blockIdx.x * blockDim.x + threadIdx.x;
  if (gid >= NB * C) return;
  int b = gid / C, c = gid - b * C;
  float4 wc = *(const float4*)&w[c * 4];
  float x0 = 0.f, x1 = 0.f, x2 = 0.f;
  size_t base = (size_t)b * NT * C + c;
  for (int t = 0; t < NT; ++t) {
    float xt = buf[base + (size_t)t * C];
    float y = wc.x * x0 + wc.y * x1 + wc.z * x2 + wc.w * xt;
    buf[base + (size_t)t * C] = siluf_(y);
    x0 = x1; x1 = x2; x2 = xt;
  }
}

// --------------------------------- per-head L2 norm (sum-of-squares) + scale -> bf16
__global__ __launch_bounds__(256) void l2norm_kernel(const float* __restrict__ in,
                                                     bf16* __restrict__ out, float scale) {
  __shared__ float red[256];
  __shared__ float rinv[NH];
  int row = blockIdx.x, tid = threadIdx.x;
  float4 xv = *(const float4*)&in[(size_t)row * KEYD + tid * 4];
  float xs[4] = {xv.x, xv.y, xv.z, xv.w};
  float ss = xs[0]*xs[0] + xs[1]*xs[1] + xs[2]*xs[2] + xs[3]*xs[3];
  red[tid] = ss; __syncthreads();
  for (int s = 32; s > 0; s >>= 1) {
    if ((tid & 63) < s) red[tid] += red[tid + s];
    __syncthreads();
  }
  if ((tid & 63) == 0) rinv[tid >> 6] = rsqrtf(red[tid] + 1e-6f) * scale;
  __syncthreads();
  float r = rinv[tid >> 6];
#pragma unroll
  for (int i = 0; i < 4; ++i)
    out[(size_t)row * KEYD + tid * 4 + i] = (bf16)(xs[i] * r);
}

// -------------------------------------- g (log-decay) and beta from h (tiny GEMVs)
__global__ __launch_bounds__(256) void gba_kernel(
    const float* __restrict__ h32, const float* __restrict__ Wa,
    const float* __restrict__ Wb, const float* __restrict__ dtb,
    const float* __restrict__ Alog, float* __restrict__ g, float* __restrict__ beta) {
  __shared__ float red[256][8];
  int row = blockIdx.x, tid = threadIdx.x;
  float a[4] = {0.f, 0.f, 0.f, 0.f}, bb[4] = {0.f, 0.f, 0.f, 0.f};
  for (int d = tid; d < ND; d += 256) {
    float hv = h32[(size_t)row * ND + d];
    float4 wa = *(const float4*)&Wa[d * 4];
    float4 wb = *(const float4*)&Wb[d * 4];
    a[0] += hv * wa.x; a[1] += hv * wa.y; a[2] += hv * wa.z; a[3] += hv * wa.w;
    bb[0] += hv * wb.x; bb[1] += hv * wb.y; bb[2] += hv * wb.z; bb[3] += hv * wb.w;
  }
#pragma unroll
  for (int i = 0; i < 4; ++i) { red[tid][i] = a[i]; red[tid][4 + i] = bb[i]; }
  __syncthreads();
  for (int s = 128; s > 0; s >>= 1) {
    if (tid < s)
#pragma unroll
      for (int i = 0; i < 8; ++i) red[tid][i] += red[tid + s][i];
    __syncthreads();
  }
  if (tid < NH) {
    float da = red[0][tid] + dtb[tid];
    float sp = (da > 20.f) ? da : log1pf(__expf(da));
    g[(size_t)row * NH + tid] = -__expf(Alog[tid]) * sp;
    beta[(size_t)row * NH + tid] = sigmoidf_(red[0][4 + tid]);
  }
}

// ------------------------------------------------- chunked gated delta-rule scan
// One workgroup per (b, h, v-block of 64).  State slice S[v][k] (64x256 f32) in LDS.
__global__ __launch_bounds__(256) void delta_chunk_kernel(
    const bf16* __restrict__ qbf, const bf16* __restrict__ kbf,
    const float* __restrict__ vglob, const float* __restrict__ gdec,
    const float* __restrict__ betag, float* __restrict__ obuf) {
  __shared__ float S[VBLK][NKD];        // state slice (v, k)
  __shared__ alignas(16) bf16 Kc[CHUNK][NKD];
  __shared__ alignas(16) bf16 Kt[NKD][CHUNK];
  __shared__ alignas(16) bf16 Qc[CHUNK][NKD];
  __shared__ float PS[CHUNK][VBLK];     // K@S then Q@S
  __shared__ float Wm[CHUNK][CHUNK];    // KK^T (masked) then QK^T (masked)
  __shared__ float Dlt[CHUNK][VBLK];
  __shared__ alignas(16) bf16 DT[VBLK][CHUNK];
  __shared__ float cum[CHUNK], ect[CHUNK], esc[CHUNK], bet[CHUNK];

  const int tid = threadIdx.x;
  const int lane = tid & 31;
  const int wid = tid >> 5;
  const int vb = blockIdx.x & (NVB - 1);
  const int h  = (blockIdx.x >> 3) & (NH - 1);
  const int b  = blockIdx.x >> 5;
  const int hi = (lane >> 4) << 3;
  const int col = lane & 15;

  for (int i = tid; i < VBLK * NKD; i += 256) (&S[0][0])[i] = 0.f;
  __syncthreads();

  for (int c = 0; c < NCHUNK; ++c) {
    const int rowbase = b * NT + c * CHUNK;
    // ---- stage K (plus transpose) and Q chunks
    for (int i = tid; i < CHUNK * NKD; i += 256) {
      int t = i >> 8, k = i & 255;
      bf16 kv = kbf[(size_t)(rowbase + t) * KEYD + h * NKD + k];
      Kc[t][k] = kv;
      Kt[k][t] = kv;
      Qc[t][k] = qbf[(size_t)(rowbase + t) * KEYD + h * NKD + k];
    }
    if (tid == 0) {
      float s = 0.f;
      for (int t = 0; t < CHUNK; ++t) {
        s += gdec[(size_t)(rowbase + t) * NH + h];
        cum[t] = s;
      }
    }
    if (tid < CHUNK) bet[tid] = betag[(size_t)(rowbase + tid) * NH + h];
    __syncthreads();
    if (tid < CHUNK) {
      ect[tid] = __expf(cum[tid]);
      esc[tid] = __expf(cum[CHUNK - 1] - cum[tid]);
    }
    __syncthreads();

    // ---- PS = K_chunk @ S0   (64x64, K'=256)
    for (int tt = wid; tt < 16; tt += 8) {
      int mt = (tt & 3) * 16, nt = (tt >> 2) * 16;
      v8f acc;
#pragma unroll
      for (int i = 0; i < 8; ++i) acc[i] = 0.f;
      for (int kb = 0; kb < NKD; kb += 32)
        acc = wmma_bf(fragA_bf(&Kc[0][0], NKD, mt, kb),
                      fragB_f32(&S[0][0], NKD, nt, kb), acc);
#pragma unroll
      for (int v = 0; v < 8; ++v) PS[mt + v + hi][nt + col] = acc[v];
    }
    __syncthreads();

    // ---- Wm = K K^T  (raw)
    for (int tt = wid; tt < 16; tt += 8) {
      int mt = (tt & 3) * 16, nt = (tt >> 2) * 16;
      v8f acc;
#pragma unroll
      for (int i = 0; i < 8; ++i) acc[i] = 0.f;
      for (int kb = 0; kb < NKD; kb += 32)
        acc = wmma_bf(fragA_bf(&Kc[0][0], NKD, mt, kb),
                      fragB_bf(&Kc[0][0], NKD, nt, kb), acc);
#pragma unroll
      for (int v = 0; v < 8; ++v) Wm[mt + v + hi][nt + col] = acc[v];
    }
    __syncthreads();
    // mask + decay + beta  ->  strictly-lower solve matrix
    for (int i = tid; i < CHUNK * CHUNK; i += 256) {
      int t = i >> 6, j = i & 63;
      Wm[t][j] = (j < t) ? bet[t] * Wm[t][j] * __expf(cum[t] - cum[j]) : 0.f;
    }
    __syncthreads();

    // ---- forward substitution: Delta = (I + Wm)^-1 @ u   (per v column)
    if (tid < VBLK) {
      int v = tid;
      for (int t = 0; t < CHUNK; ++t) {
        float u = bet[t] * (vglob[((size_t)(rowbase + t) * NH + h) * NVD + vb * VBLK + v]
                            - ect[t] * PS[t][v]);
        for (int j = 0; j < t; ++j) u -= Wm[t][j] * Dlt[j][v];
        Dlt[t][v] = u;
        DT[v][t] = (bf16)u;
      }
    }
    __syncthreads();

    // ---- PS = Q_chunk @ S0
    for (int tt = wid; tt < 16; tt += 8) {
      int mt = (tt & 3) * 16, nt = (tt >> 2) * 16;
      v8f acc;
#pragma unroll
      for (int i = 0; i < 8; ++i) acc[i] = 0.f;
      for (int kb = 0; kb < NKD; kb += 32)
        acc = wmma_bf(fragA_bf(&Qc[0][0], NKD, mt, kb),
                      fragB_f32(&S[0][0], NKD, nt, kb), acc);
#pragma unroll
      for (int v = 0; v < 8; ++v) PS[mt + v + hi][nt + col] = acc[v];
    }
    __syncthreads();

    // ---- Wm = Q K^T  (raw), then inclusive mask + decay
    for (int tt = wid; tt < 16; tt += 8) {
      int mt = (tt & 3) * 16, nt = (tt >> 2) * 16;
      v8f acc;
#pragma unroll
      for (int i = 0; i < 8; ++i) acc[i] = 0.f;
      for (int kb = 0; kb < NKD; kb += 32)
        acc = wmma_bf(fragA_bf(&Qc[0][0], NKD, mt, kb),
                      fragB_bf(&Kc[0][0], NKD, nt, kb), acc);
#pragma unroll
      for (int v = 0; v < 8; ++v) Wm[mt + v + hi][nt + col] = acc[v];
    }
    __syncthreads();
    for (int i = tid; i < CHUNK * CHUNK; i += 256) {
      int t = i >> 6, j = i & 63;
      Wm[t][j] = (j <= t) ? Wm[t][j] * __expf(cum[t] - cum[j]) : 0.f;
    }
    __syncthreads();

    // ---- O = Wm @ Delta + diag(exp(cum)) @ PS   -> global
    for (int tt = wid; tt < 16; tt += 8) {
      int mt = (tt & 3) * 16, nt = (tt >> 2) * 16;
      v8f acc;
#pragma unroll
      for (int i = 0; i < 8; ++i) acc[i] = 0.f;
      for (int kb = 0; kb < CHUNK; kb += 32)
        acc = wmma_bf(fragA_f32(&Wm[0][0], CHUNK, mt, kb),
                      fragB_bf(&DT[0][0], CHUNK, nt, kb), acc);
#pragma unroll
      for (int v = 0; v < 8; ++v) {
        int t = mt + v + hi, vc = nt + col;
        float o = acc[v] + ect[t] * PS[t][vc];
        obuf[((size_t)(rowbase + t) * NH + h) * NVD + vb * VBLK + vc] = o;
      }
    }
    __syncthreads();

    // ---- state update: S = exp(cum63)*S + (Delta^T . esc) @ K_chunk
    const float sdec = __expf(cum[CHUNK - 1]);
    for (int tt = wid; tt < 64; tt += 8) {
      int mt = (tt & 3) * 16, nt = (tt >> 2) * 16;
      v8f acc;
#pragma unroll
      for (int i = 0; i < 8; ++i) acc[i] = 0.f;
      for (int kb = 0; kb < CHUNK; kb += 32)
        acc = wmma_bf(fragA_bf_scaled(&DT[0][0], CHUNK, mt, kb, esc),
                      fragB_bf(&Kt[0][0], CHUNK, nt, kb), acc);
#pragma unroll
      for (int v = 0; v < 8; ++v) {
        int vv = mt + v + hi, kk = nt + col;
        S[vv][kk] = sdec * S[vv][kk] + acc[v];
      }
    }
    __syncthreads();
  }
}

// ------------------------------- gated RMSNorm of o with SiLU(gate), -> bf16 rows
__global__ __launch_bounds__(256) void onorm_gate_kernel(
    const float* __restrict__ obuf, const float* __restrict__ onw,
    const float* __restrict__ gate, bf16* __restrict__ ogbf) {
  __shared__ float red[256];
  __shared__ float rinv[NH];
  int row = blockIdx.x, tid = threadIdx.x;
  float xs[8];
  float ss = 0.f;
#pragma unroll
  for (int i = 0; i < 8; ++i) {
    xs[i] = obuf[(size_t)row * VALD + tid * 8 + i];
    ss += xs[i] * xs[i];
  }
  red[tid] = ss; __syncthreads();
  for (int s = 32; s > 0; s >>= 1) {
    if ((tid & 63) < s) red[tid] += red[tid + s];
    __syncthreads();
  }
  if ((tid & 63) == 0) rinv[tid >> 6] = rsqrtf(red[tid] * (1.f / NVD) + 1e-6f);
  __syncthreads();
  float r = rinv[tid >> 6];
#pragma unroll
  for (int i = 0; i < 8; ++i) {
    int vd = (tid & 63) * 8 + i;
    float gv = gate[(size_t)row * VALD + tid * 8 + i];
    float v = xs[i] * r * onw[vd] * siluf_(gv);
    ogbf[(size_t)row * VALD + tid * 8 + i] = (bf16)v;
  }
}

// ------------------------------------ residual add + LN2 -> x1 (f32) + h2 (bf16)
__global__ __launch_bounds__(256) void resid_ln2_kernel(
    const float* __restrict__ x, const float* __restrict__ enc,
    const float* __restrict__ g2, const float* __restrict__ b2ln,
    float* __restrict__ x1, bf16* __restrict__ h2bf) {
  __shared__ float red[256];
  int row = blockIdx.x, tid = threadIdx.x;
  float xs[4];
#pragma unroll
  for (int i = 0; i < 4; ++i) {
    int c = tid * 4 + i;
    xs[i] = x[(size_t)row * ND + c] + enc[(size_t)row * ND + c];
    x1[(size_t)row * ND + c] = xs[i];
  }
  float s = xs[0] + xs[1] + xs[2] + xs[3];
  float mean = block_sum256(s, red) * (1.f / ND);
  float s2 = 0.f;
#pragma unroll
  for (int i = 0; i < 4; ++i) { float d = xs[i] - mean; s2 += d * d; }
  float var = block_sum256(s2, red) * (1.f / ND);
  float inv = rsqrtf(var + 1e-5f);
#pragma unroll
  for (int i = 0; i < 4; ++i) {
    int c = tid * 4 + i;
    h2bf[(size_t)row * ND + c] = (bf16)((xs[i] - mean) * inv * g2[c] + b2ln[c]);
  }
}

// --------------------------------------------- out = gelu(y + b2) + x1 (exact gelu)
__global__ void final_gelu_kernel(const float* __restrict__ y,
                                  const float* __restrict__ bias,
                                  const float* __restrict__ x1,
                                  float* __restrict__ out) {
  size_t total = (size_t)MROWS * ND;
  for (size_t i = blockIdx.x * blockDim.x + threadIdx.x; i < total;
       i += (size_t)gridDim.x * blockDim.x) {
    float z = y[i] + bias[i & (ND - 1)];
    float ge = 0.5f * z * (1.f + erff(z * 0.70710678118654752f));
    out[i] = ge + x1[i];
  }
}

// ================================================================== host driver
extern "C" void kernel_launch(void* const* d_in, const int* in_sizes, int n_in,
                              void* d_out, int out_size, void* d_ws, size_t ws_size,
                              hipStream_t stream) {
  const float* x     = (const float*)d_in[0];
  const float* ln1g  = (const float*)d_in[1];
  const float* ln1b  = (const float*)d_in[2];
  const float* ln2g  = (const float*)d_in[3];
  const float* ln2b  = (const float*)d_in[4];
  const float* rmsw  = (const float*)d_in[5];
  const float* Wq    = (const float*)d_in[6];
  const float* Wk    = (const float*)d_in[7];
  const float* Wv    = (const float*)d_in[8];
  const float* cqw   = (const float*)d_in[9];
  const float* ckw   = (const float*)d_in[10];
  const float* cvw   = (const float*)d_in[11];
  const float* Wa    = (const float*)d_in[12];
  const float* dtb   = (const float*)d_in[13];
  const float* Alog  = (const float*)d_in[14];
  const float* Wb    = (const float*)d_in[15];
  const float* Wg    = (const float*)d_in[16];
  const float* onw   = (const float*)d_in[17];
  const float* Wo    = (const float*)d_in[18];
  const float* W2    = (const float*)d_in[19];
  const float* b2    = (const float*)d_in[20];
  float* out = (float*)d_out;

  char* wsp = (char*)d_ws;
  size_t off = 0;
  auto alloc = [&](size_t bytes) -> void* {
    void* p = wsp + off;
    off += (bytes + 255) & ~(size_t)255;
    return p;
  };
  bf16*  WqT  = (bf16*)alloc((size_t)KEYD * ND * 2);
  bf16*  WkT  = (bf16*)alloc((size_t)KEYD * ND * 2);
  bf16*  WvT  = (bf16*)alloc((size_t)VALD * ND * 2);
  bf16*  WgT  = (bf16*)alloc((size_t)VALD * ND * 2);
  bf16*  WoT  = (bf16*)alloc((size_t)ND * VALD * 2);
  bf16*  W2T  = (bf16*)alloc((size_t)ND * ND * 2);
  float* h32  = (float*)alloc((size_t)MROWS * ND * 4);
  bf16*  hbf  = (bf16*)alloc((size_t)MROWS * ND * 2);
  float* qbuf = (float*)alloc((size_t)MROWS * KEYD * 4);
  float* kbuf = (float*)alloc((size_t)MROWS * KEYD * 4);
  float* vbuf = (float*)alloc((size_t)MROWS * VALD * 4);
  float* gate = (float*)alloc((size_t)MROWS * VALD * 4);
  bf16*  qbf  = (bf16*)alloc((size_t)MROWS * KEYD * 2);
  bf16*  kbf  = (bf16*)alloc((size_t)MROWS * KEYD * 2);
  float* gdec = (float*)alloc((size_t)MROWS * NH * 4);
  float* betb = (float*)alloc((size_t)MROWS * NH * 4);
  float* obuf = (float*)alloc((size_t)MROWS * VALD * 4);
  bf16*  ogbf = (bf16*)alloc((size_t)MROWS * VALD * 2);
  float* encb = (float*)alloc((size_t)MROWS * ND * 4);
  float* x1b  = (float*)alloc((size_t)MROWS * ND * 4);
  bf16*  h2bf = (bf16*)alloc((size_t)MROWS * ND * 2);
  float* yb   = (float*)alloc((size_t)MROWS * ND * 4);
  (void)ws_size; (void)in_sizes; (void)n_in; (void)out_size;

  // weights -> transposed bf16
  transpose_bf16_kernel<<<2048, 256, 0, stream>>>(Wq, WqT, ND, KEYD);
  transpose_bf16_kernel<<<2048, 256, 0, stream>>>(Wk, WkT, ND, KEYD);
  transpose_bf16_kernel<<<2048, 256, 0, stream>>>(Wv, WvT, ND, VALD);
  transpose_bf16_kernel<<<2048, 256, 0, stream>>>(Wg, WgT, ND, VALD);
  transpose_bf16_kernel<<<2048, 256, 0, stream>>>(Wo, WoT, VALD, ND);
  transpose_bf16_kernel<<<2048, 256, 0, stream>>>(W2, W2T, ND, ND);

  // LN1 + RMSNorm
  norm1_kernel<<<MROWS, 256, 0, stream>>>(x, ln1g, ln1b, rmsw, h32, hbf);

  // projections (WMMA GEMMs)
  gemm_bf16_kernel<<<dim3(MROWS / 128, KEYD / 128), 256, 0, stream>>>(hbf, WqT, qbuf, MROWS, KEYD, ND);
  gemm_bf16_kernel<<<dim3(MROWS / 128, KEYD / 128), 256, 0, stream>>>(hbf, WkT, kbuf, MROWS, KEYD, ND);
  gemm_bf16_kernel<<<dim3(MROWS / 128, VALD / 128), 256, 0, stream>>>(hbf, WvT, vbuf, MROWS, VALD, ND);
  gemm_bf16_kernel<<<dim3(MROWS / 128, VALD / 128), 256, 0, stream>>>(hbf, WgT, gate, MROWS, VALD, ND);

  // decay / beta gates
  gba_kernel<<<MROWS, 256, 0, stream>>>(h32, Wa, Wb, dtb, Alog, gdec, betb);

  // causal conv + SiLU (in place)
  conv_silu_kernel<<<(NB * KEYD + 255) / 256, 256, 0, stream>>>(qbuf, cqw, KEYD);
  conv_silu_kernel<<<(NB * KEYD + 255) / 256, 256, 0, stream>>>(kbuf, ckw, KEYD);
  conv_silu_kernel<<<(NB * VALD + 255) / 256, 256, 0, stream>>>(vbuf, cvw, VALD);

  // per-head L2 norm; q additionally scaled by Kd^-0.5
  l2norm_kernel<<<MROWS, 256, 0, stream>>>(qbuf, qbf, 0.0625f);
  l2norm_kernel<<<MROWS, 256, 0, stream>>>(kbuf, kbf, 1.0f);

  // chunked gated delta rule (WMMA-heavy scan)
  delta_chunk_kernel<<<NB * NH * NVB, 256, 0, stream>>>(qbf, kbf, vbuf, gdec, betb, obuf);

  // gated RMSNorm output
  onorm_gate_kernel<<<MROWS, 256, 0, stream>>>(obuf, onw, gate, ogbf);

  // o_proj
  gemm_bf16_kernel<<<dim3(MROWS / 128, ND / 128), 256, 0, stream>>>(ogbf, WoT, encb, MROWS, ND, VALD);

  // residual + LN2
  resid_ln2_kernel<<<MROWS, 256, 0, stream>>>(x, encb, ln2g, ln2b, x1b, h2bf);

  // stage-2 GEMM
  gemm_bf16_kernel<<<dim3(MROWS / 128, ND / 128), 256, 0, stream>>>(h2bf, W2T, yb, MROWS, ND, ND);

  // GELU + residual
  final_gelu_kernel<<<2048, 256, 0, stream>>>(yb, b2, x1b, out);
}
